// TemporalAttention_15436112462270
// MI455X (gfx1250) — compile-verified, matrix operand order checked by reading, not run on hardware
//
#include <hip/hip_runtime.h>
#include <math.h>

#define HID 128
#define NHD 4
#define DHD 32

typedef __bf16 bf16_t;
typedef __attribute__((ext_vector_type(16))) __bf16 v16bf;
typedef __attribute__((ext_vector_type(8)))  __bf16 v8bf;
typedef __attribute__((ext_vector_type(8)))  float  v8f;

// ---------------- helpers ----------------

__device__ inline bf16_t f2bf(float f) {
  unsigned u = __builtin_bit_cast(unsigned, f);
  u += 0x7fffu + ((u >> 16) & 1u);           // round-to-nearest-even
  unsigned short h = (unsigned short)(u >> 16);
  return __builtin_bit_cast(bf16_t, h);
}

// monotone float <-> uint mapping for atomicMax on floats
__device__ inline unsigned ford(float f) {
  unsigned u = __builtin_bit_cast(unsigned, f);
  return (u & 0x80000000u) ? ~u : (u | 0x80000000u);
}
__device__ inline float deford(unsigned u) {
  unsigned v = (u & 0x80000000u) ? (u & 0x7fffffffu) : ~u;
  return __builtin_bit_cast(float, v);
}

enum {
  MODE_BIAS = 0,      // x + b
  MODE_SCALE = 1,     // (x + b) * scale
  MODE_SILU = 2,      // silu(x + b)
  MODE_SIGMOID = 3,   // sigmoid(x + b)
  MODE_GATE_RES = 4,  // aux1 + aux0 * (x + b)   (h_corr = h_prev + g*out)
  MODE_ADD_AUX = 5    // aux0 + (x + b)          (fused = h_corr + mlp)
};

template <int MODE>
__device__ inline float apply_mode(float x, float scale,
                                   const float* __restrict__ aux0,
                                   const float* __restrict__ aux1, size_t idx) {
  if constexpr (MODE == MODE_SCALE)    return x * scale;
  else if constexpr (MODE == MODE_SILU)    return x / (1.0f + __expf(-x));
  else if constexpr (MODE == MODE_SIGMOID) return 1.0f / (1.0f + __expf(-x));
  else if constexpr (MODE == MODE_GATE_RES) return aux1[idx] + aux0[idx] * x;
  else if constexpr (MODE == MODE_ADD_AUX)  return aux0[idx] + x;
  else return x;
}

// ---------------- bf16 WMMA GEMM ----------------
// C[M x 128] = epilogue( [A0 | A1][M x KTOT] @ W[KTOT x 128] + bias )
// A0/A1: row-major bf16, 128 cols each. Wp: pre-packed B-fragment-major bf16.
// One wave = one 16x16 tile; 256-thread block = 16-row strip x 128 cols.
// MODE/output formats are compile-time -> straight-line epilogue.
template <int KTOT, bool CONCAT, int MODE, bool WF32, bool WBF16>
__global__ __launch_bounds__(256)
void gemm_wmma_bf16(const bf16_t* __restrict__ A0, const bf16_t* __restrict__ A1,
                    const bf16_t* __restrict__ Wp, const float* __restrict__ bias,
                    float* __restrict__ Cf, bf16_t* __restrict__ Cb,
                    int Nrows, float scale,
                    const float* __restrict__ aux0, const float* __restrict__ aux1) {
  constexpr int KB = KTOT >> 5;             // number of 32-wide K blocks
  const int lane    = threadIdx.x & 31;
  const int wave    = threadIdx.x >> 5;     // n-tile id (0..7)
  const int rowBase = blockIdx.x << 4;
  const int l15     = lane & 15;
  const int khalf   = (lane >> 4) << 3;     // 0 or 8
  const int ncol    = (wave << 4) + l15;

  int mload = rowBase + l15;                // clamp keeps EXEC all-ones at WMMA
  if (mload >= Nrows) mload = Nrows - 1;
  const bf16_t* a0row = A0 + (size_t)mload * HID;
  const bf16_t* a1row = CONCAT ? (A1 + (size_t)mload * HID) : a0row;
  const bf16_t* wbase = Wp + (((size_t)wave * KB * 32 + lane) << 4);

  v8f acc = {};

#pragma unroll
  for (int kb = 0; kb < KTOT; kb += 32) {
    const int kLo = kb + khalf;             // multiple of 8 -> 16B aligned
    const int kHi = kb + 16 + khalf;

    const bf16_t* pLo;
    const bf16_t* pHi;
    if (CONCAT) {
      pLo = (kLo < HID) ? (a0row + kLo) : (a1row + (kLo - HID));
      pHi = (kHi < HID) ? (a0row + kHi) : (a1row + (kHi - HID));
    } else {
      pLo = a0row + kLo;
      pHi = a0row + kHi;
    }
    v8bf alo = *(const v8bf*)pLo;
    v8bf ahi = *(const v8bf*)pHi;
    v16bf afrag = __builtin_shufflevector(alo, ahi,
        0, 1, 2, 3, 4, 5, 6, 7, 8, 9, 10, 11, 12, 13, 14, 15);

    const bf16_t* bp = wbase + (((size_t)(kb >> 5) * 32) << 4);
    v8bf blo = *(const v8bf*)bp;
    v8bf bhi = *(const v8bf*)(bp + 8);
    v16bf bfrag = __builtin_shufflevector(blo, bhi,
        0, 1, 2, 3, 4, 5, 6, 7, 8, 9, 10, 11, 12, 13, 14, 15);

    if (kb + 32 < KTOT)
      __builtin_prefetch(bp + 512, 0, 1);   // next K-slab of packed W

    acc = __builtin_amdgcn_wmma_f32_16x16x32_bf16(
        false, afrag, false, bfrag, (short)0, acc, false, false);
  }

  // epilogue: lane owns column ncol, rows mBase..mBase+7 (stores coalesce
  // across lanes: fixed row, consecutive columns per half-wave)
  const int mBase = rowBase + ((lane >> 4) << 3);
  const float b = bias[ncol];
  if (rowBase + 16 <= Nrows) {              // full tile: no per-row guards
#pragma unroll
    for (int r = 0; r < 8; ++r) {
      const size_t idx = (size_t)(mBase + r) * HID + ncol;
      const float res = apply_mode<MODE>(acc[r] + b, scale, aux0, aux1, idx);
      if constexpr (WF32)  Cf[idx] = res;
      if constexpr (WBF16) Cb[idx] = f2bf(res);
    }
  } else {
#pragma unroll
    for (int r = 0; r < 8; ++r) {
      const int m = mBase + r;
      if (m >= Nrows) continue;
      const size_t idx = (size_t)m * HID + ncol;
      const float res = apply_mode<MODE>(acc[r] + b, scale, aux0, aux1, idx);
      if constexpr (WF32)  Cf[idx] = res;
      if constexpr (WBF16) Cb[idx] = f2bf(res);
    }
  }
}

// ---------------- data movement / packing ----------------

__global__ void zero_u32(unsigned* __restrict__ p, size_t n) {
  size_t i = (size_t)blockIdx.x * blockDim.x + threadIdx.x;
  if (i < n) p[i] = 0u;
}

__global__ void cvt_f32_bf16(const float* __restrict__ src, bf16_t* __restrict__ dst, size_t n) {
  size_t i = (size_t)blockIdx.x * blockDim.x + threadIdx.x;
  if (i < n) dst[i] = f2bf(src[i]);
}

// Pack W[KTOT x 128] (fp32 row-major) into B-fragment-major bf16:
// P[(((nt*KB + kb32)*32 + lane)*16 + e] = bf16(W[k][n])
//   n = nt*16 + (lane&15); khalf = (lane>>4)*8
//   k = kb32*32 + (e<8 ? khalf+e : 16+khalf+(e-8))
__global__ void pack_w_bf16(const float* __restrict__ W, bf16_t* __restrict__ P, int KTOT) {
  const int t = blockIdx.x * blockDim.x + threadIdx.x;
  const int total = KTOT * HID;
  if (t >= total) return;
  const int KB = KTOT >> 5;
  const int e    = t & 15;
  const int lane = (t >> 4) & 31;
  const int rest = t >> 9;
  const int kb32 = rest % KB;
  const int nt   = rest / KB;
  const int n = (nt << 4) + (lane & 15);
  const int khalf = (lane >> 4) << 3;
  const int k = (kb32 << 5) + ((e < 8) ? (khalf + e) : (16 + khalf + (e - 8)));
  P[t] = f2bf(W[(size_t)k * HID + n]);
}

// ---------------- sparse attention kernels ----------------

__global__ void edge_scores_kernel(const float* __restrict__ q, const float* __restrict__ k,
                                   const int* __restrict__ rows, const int* __restrict__ cols,
                                   float* __restrict__ scores, unsigned* __restrict__ rmax,
                                   int Ee) {
  int e = blockIdx.x * blockDim.x + threadIdx.x;
  if (e >= Ee) return;
  const int r = rows[e], c = cols[e];
  const float4* qr = (const float4*)(q + (size_t)r * HID);
  const float4* kc = (const float4*)(k + (size_t)c * HID);
  float ax = 0.f, ay = 0.f, az = 0.f, aw = 0.f;
#pragma unroll
  for (int d = 0; d < DHD; ++d) {
    float4 a = qr[d], b = kc[d];
    ax += a.x * b.x; ay += a.y * b.y; az += a.z * b.z; aw += a.w * b.w;
  }
  float4 s; s.x = ax; s.y = ay; s.z = az; s.w = aw;
  ((float4*)scores)[e] = s;
  unsigned* rm = rmax + (size_t)r * NHD;
  atomicMax(rm + 0, ford(ax));
  atomicMax(rm + 1, ford(ay));
  atomicMax(rm + 2, ford(az));
  atomicMax(rm + 3, ford(aw));
}

__global__ void edge_exp_kernel(float* __restrict__ scores, const unsigned* __restrict__ rmax,
                                const int* __restrict__ rows, float* __restrict__ denom,
                                int Ee) {
  int t = blockIdx.x * blockDim.x + threadIdx.x;
  if (t >= Ee * NHD) return;
  const int e = t >> 2, h = t & 3;
  const int r = rows[e];
  float m = deford(rmax[(size_t)r * NHD + h]);
  if (!isfinite(m)) m = 0.0f;
  const float v = __expf(scores[t] - m);
  scores[t] = v;
  atomicAdd(&denom[(size_t)r * NHD + h], v);
}

__global__ void edge_spmm_kernel(const float* __restrict__ ex, const float* __restrict__ denom,
                                 const float* __restrict__ v,
                                 const int* __restrict__ rows, const int* __restrict__ cols,
                                 float* __restrict__ y, int Ee) {
  int t = blockIdx.x * blockDim.x + threadIdx.x;
  if (t >= Ee * DHD) return;
  const int e = t >> 5, d = t & 31;
  const int r = rows[e], c = cols[e];
  const float4 vv = ((const float4*)(v + (size_t)c * HID))[d];
  const float4 ev = ((const float4*)ex)[e];
  const float4 dn = ((const float4*)denom)[r];
  const float a0 = ev.x / (dn.x > 0.f ? dn.x : 1.0f);
  const float a1 = ev.y / (dn.y > 0.f ? dn.y : 1.0f);
  const float a2 = ev.z / (dn.z > 0.f ? dn.z : 1.0f);
  const float a3 = ev.w / (dn.w > 0.f ? dn.w : 1.0f);
  float* yr = y + (size_t)r * HID + d * 4;
  atomicAdd(yr + 0, a0 * vv.x);
  atomicAdd(yr + 1, a1 * vv.y);
  atomicAdd(yr + 2, a2 * vv.z);
  atomicAdd(yr + 3, a3 * vv.w);
}

// ---------------- launch ----------------

extern "C" void kernel_launch(void* const* d_in, const int* in_sizes, int n_in,
                              void* d_out, int out_size, void* d_ws, size_t ws_size,
                              hipStream_t stream) {
  const float* h_prev = (const float*)d_in[0];
  const float* h_pred = (const float*)d_in[1];
  const float* Wq = (const float*)d_in[2];   const float* bq = (const float*)d_in[3];
  const float* Wk = (const float*)d_in[4];   const float* bk = (const float*)d_in[5];
  const float* Wv = (const float*)d_in[6];   const float* bv = (const float*)d_in[7];
  const float* Wo = (const float*)d_in[8];   const float* bo = (const float*)d_in[9];
  const float* Wg1 = (const float*)d_in[10]; const float* bg1 = (const float*)d_in[11];
  const float* Wg2 = (const float*)d_in[12]; const float* bg2 = (const float*)d_in[13];
  const float* Wm1 = (const float*)d_in[14]; const float* bm1 = (const float*)d_in[15];
  const float* Wm2 = (const float*)d_in[16]; const float* bm2 = (const float*)d_in[17];
  const int* rows = (const int*)d_in[18];
  const int* cols = (const int*)d_in[19];

  const int Nn = in_sizes[0] / HID;
  const int Ee = in_sizes[18];
  const size_t nh = (size_t)Nn * HID;

  // ---- workspace layout: fp32 region, then bf16 region ----
  float* q_ = (float*)d_ws;
  float* k_ = q_ + nh;
  float* v_ = k_ + nh;
  float* y_ = v_ + nh;
  float* g_ = y_ + nh;
  float* hc = g_ + nh;
  float* sc = hc + nh;                               // E*NH scores (then ex, in place)
  unsigned* rmax = (unsigned*)(sc + (size_t)Ee * NHD);
  float* den = (float*)(rmax + (size_t)Nn * NHD);

  bf16_t* hpred_bf = (bf16_t*)(den + (size_t)Nn * NHD);
  bf16_t* hprev_bf = hpred_bf + nh;
  bf16_t* t1_bf    = hprev_bf + nh;
  bf16_t* y_bf     = t1_bf + nh;
  bf16_t* hc_bf    = y_bf + nh;
  bf16_t* wq_p  = hc_bf + nh;          // 128*128 each
  bf16_t* wk_p  = wq_p + HID * HID;
  bf16_t* wv_p  = wk_p + HID * HID;
  bf16_t* wo_p  = wv_p + HID * HID;
  bf16_t* wg2_p = wo_p + HID * HID;
  bf16_t* wm2_p = wg2_p + HID * HID;
  bf16_t* wg1_p = wm2_p + HID * HID;   // 256*128 each
  bf16_t* wm1_p = wg1_p + 2 * HID * HID;

  const dim3 blk(256);
  const int mtiles = (Nn + 15) / 16;
  const float qscale = rsqrtf((float)DHD);
  const unsigned gcvt = (unsigned)((nh + 255) / 256);
  const unsigned gpw1 = (HID * HID + 255) / 256;
  const unsigned gpw2 = (2 * HID * HID + 255) / 256;

  // zero y and rowmax+denom
  zero_u32<<<gcvt, blk, 0, stream>>>((unsigned*)y_, nh);
  zero_u32<<<(unsigned)(((size_t)Nn * NHD * 2 + 255) / 256), blk, 0, stream>>>(
      rmax, (size_t)Nn * NHD * 2);

  // activations -> bf16, weights -> packed bf16
  cvt_f32_bf16<<<gcvt, blk, 0, stream>>>(h_pred, hpred_bf, nh);
  cvt_f32_bf16<<<gcvt, blk, 0, stream>>>(h_prev, hprev_bf, nh);
  pack_w_bf16<<<gpw1, blk, 0, stream>>>(Wq, wq_p, HID);
  pack_w_bf16<<<gpw1, blk, 0, stream>>>(Wk, wk_p, HID);
  pack_w_bf16<<<gpw1, blk, 0, stream>>>(Wv, wv_p, HID);
  pack_w_bf16<<<gpw1, blk, 0, stream>>>(Wo, wo_p, HID);
  pack_w_bf16<<<gpw1, blk, 0, stream>>>(Wg2, wg2_p, HID);
  pack_w_bf16<<<gpw1, blk, 0, stream>>>(Wm2, wm2_p, HID);
  pack_w_bf16<<<gpw2, blk, 0, stream>>>(Wg1, wg1_p, 2 * HID);
  pack_w_bf16<<<gpw2, blk, 0, stream>>>(Wm1, wm1_p, 2 * HID);

  // q = (h_pred@Wq+bq)/sqrt(32) ; k = h_prev@Wk+bk ; v = h_pred@Wv+bv
  gemm_wmma_bf16<HID, false, MODE_SCALE, true, false><<<mtiles, blk, 0, stream>>>(
      hpred_bf, nullptr, wq_p, bq, q_, nullptr, Nn, qscale, nullptr, nullptr);
  gemm_wmma_bf16<HID, false, MODE_BIAS, true, false><<<mtiles, blk, 0, stream>>>(
      hprev_bf, nullptr, wk_p, bk, k_, nullptr, Nn, 1.f, nullptr, nullptr);
  gemm_wmma_bf16<HID, false, MODE_BIAS, true, false><<<mtiles, blk, 0, stream>>>(
      hpred_bf, nullptr, wv_p, bv, v_, nullptr, Nn, 1.f, nullptr, nullptr);

  // sparse attention
  edge_scores_kernel<<<(Ee + 255) / 256, blk, 0, stream>>>(q_, k_, rows, cols, sc, rmax, Ee);
  edge_exp_kernel<<<(unsigned)(((size_t)Ee * NHD + 255) / 256), blk, 0, stream>>>(
      sc, rmax, rows, den, Ee);
  edge_spmm_kernel<<<(unsigned)(((size_t)Ee * DHD + 255) / 256), blk, 0, stream>>>(
      sc, den, v_, rows, cols, y_, Ee);
  cvt_f32_bf16<<<gcvt, blk, 0, stream>>>(y_, y_bf, nh);

  // gate branch: t1 = silu([h_pred|h_prev]@Wg1+bg1) ; g = sigmoid(t1@Wg2+bg2)
  gemm_wmma_bf16<2 * HID, true, MODE_SILU, false, true><<<mtiles, blk, 0, stream>>>(
      hpred_bf, hprev_bf, wg1_p, bg1, nullptr, t1_bf, Nn, 1.f, nullptr, nullptr);
  gemm_wmma_bf16<HID, false, MODE_SIGMOID, true, false><<<mtiles, blk, 0, stream>>>(
      t1_bf, nullptr, wg2_p, bg2, g_, nullptr, Nn, 1.f, nullptr, nullptr);
  // h_corr = h_prev + g * (y@Wo + bo)   (fp32 + bf16 copies)
  gemm_wmma_bf16<HID, false, MODE_GATE_RES, true, true><<<mtiles, blk, 0, stream>>>(
      y_bf, nullptr, wo_p, bo, hc, hc_bf, Nn, 1.f, g_, h_prev);
  // fused = h_corr + silu([h_corr|h_prev]@Wm1+bm1)@Wm2 + bm2
  gemm_wmma_bf16<2 * HID, true, MODE_SILU, false, true><<<mtiles, blk, 0, stream>>>(
      hc_bf, hprev_bf, wm1_p, bm1, nullptr, t1_bf, Nn, 1.f, nullptr, nullptr);
  gemm_wmma_bf16<HID, false, MODE_ADD_AUX, true, false><<<mtiles, blk, 0, stream>>>(
      t1_bf, nullptr, wm2_p, bm2, (float*)d_out, nullptr, Nn, 1.f, hc, nullptr);
}
